// NodeModelWithGlobal_5428838662515
// MI455X (gfx1250) — matrix operand order repaired and checked by reading, not compiled, not run on hardware
//
#include <hip/hip_runtime.h>
#include <hip/hip_bf16.h>

// ---------------- problem constants (from reference) ----------------
#define N_NODES   50000
#define N_EDGES   800000
#define NODE_IN   256
#define EDGE_OUT  64
#define GLOBAL_IN 128
#define NODE_OUT  256
#define HIDDEN    512
#define IN_DIM    (NODE_IN + EDGE_OUT + GLOBAL_IN)   // 448
#define KC1       (IN_DIM / 32)                      // 14 K-chunks, GEMM1
#define KC2       (HIDDEN / 32)                      // 16 K-chunks, GEMM2

// ---------------- tile / LDS layout ----------------
#define M_TILE    64          // nodes per workgroup
#define A_STRIDE  456         // 448 + 8 pad (228 dwords; 36 mod 64 lane step -> conflict-free)
#define H_STRIDE  520         // 512 + 8 pad (260 dwords; 4 mod 64 lane step)
#define W_STRIDE  40          // 32 + 8 pad  (20 dword lane step)
#define CH1_ELEMS (HIDDEN * W_STRIDE)                // 20480 bf16 per GEMM1 chunk
#define CH2_ELEMS (NODE_OUT * W_STRIDE)              // 10240 bf16 per GEMM2 chunk

// workspace byte offsets
#define SUMS_FLOATS   ((size_t)N_NODES * EDGE_OUT)
#define W1P_OFF_B     ((SUMS_FLOATS + N_NODES) * 4)                    // 13,000,000 (16B aligned)
#define W1P_BYTES     ((size_t)KC1 * CH1_ELEMS * 2)                    // 573,440
#define W2P_OFF_B     (W1P_OFF_B + W1P_BYTES)                          // 13,573,440 (16B aligned)

typedef __attribute__((ext_vector_type(4)))  __bf16 v4bf;
typedef __attribute__((ext_vector_type(8)))  __bf16 v8bf;
typedef __attribute__((ext_vector_type(16))) __bf16 v16bf;
typedef __attribute__((ext_vector_type(4)))  float  v4f;
typedef __attribute__((ext_vector_type(8)))  float  v8f;
typedef __attribute__((ext_vector_type(4)))  int    v4i;

typedef __attribute__((address_space(1))) v4i* gv4i_ptr;   // global int4*
typedef __attribute__((address_space(3))) v4i* lv4i_ptr;   // LDS int4*

static __device__ __forceinline__ v16bf cat8(v8bf lo, v8bf hi) {
    return __builtin_shufflevector(lo, hi, 0,1,2,3,4,5,6,7,8,9,10,11,12,13,14,15);
}

// -------- gfx1250 async global->LDS copy (ASYNCcnt path), with safe fallback --------
static __device__ __forceinline__ void wait_async0() {
#if __has_builtin(__builtin_amdgcn_s_wait_asynccnt)
    __builtin_amdgcn_s_wait_asynccnt(0);
#else
    asm volatile("s_wait_asynccnt 0x0" ::: "memory");
#endif
}

// issue copy of n128 * 16 bytes (no wait), 256 threads cooperating
static __device__ __forceinline__ void copy16_issue(__bf16* dst, const __bf16* src,
                                                    int n128, int tid) {
#if __has_builtin(__builtin_amdgcn_global_load_async_to_lds_b128)
    gv4i_ptr g = (gv4i_ptr)(src);
    lv4i_ptr l = (lv4i_ptr)(dst);
    for (int i = tid; i < n128; i += 256)
        __builtin_amdgcn_global_load_async_to_lds_b128(g + i, l + i, 0, 0);
#else
    const v8bf* s = (const v8bf*)src;
    v8bf*       d = (v8bf*)dst;
    for (int i = tid; i < n128; i += 256) d[i] = s[i];
#endif
}

// B-fragment fetch from an LDS weight chunk laid out [n][W_STRIDE]
static __device__ __forceinline__ v16bf loadB(const __bf16* sWc, int n, int l16, int lhalf) {
    const __bf16* bP = sWc + (n * 16 + l16) * W_STRIDE + lhalf * 8;
    return cat8(*(const v8bf*)bP, *(const v8bf*)(bP + 16));
}

// ---------------- K0: pack weights to bf16, transposed to [kc][n][W_STRIDE] ----------------
__global__ __launch_bounds__(256) void pack_weights_kernel(
        const float* __restrict__ W1, const float* __restrict__ W2,
        __bf16* __restrict__ W1P, __bf16* __restrict__ W2P)
{
    int i = blockIdx.x * 256 + threadIdx.x;
    if (i < IN_DIM * HIDDEN) {                 // coalesced over n
        int k = i / HIDDEN, n = i - k * HIDDEN;
        W1P[(size_t)(k >> 5) * CH1_ELEMS + n * W_STRIDE + (k & 31)] = (__bf16)W1[i];
    }
    if (i < HIDDEN * NODE_OUT) {
        int k = i / NODE_OUT, n = i - k * NODE_OUT;
        W2P[(size_t)(k >> 5) * CH2_ELEMS + n * W_STRIDE + (k & 31)] = (__bf16)W2[i];
    }
}

// ---------------- K1: zero sums + counts ----------------
__global__ void zero_ws_kernel(float* __restrict__ p, int n) {
    int i = blockIdx.x * blockDim.x + threadIdx.x;
    if (i < n) p[i] = 0.0f;
}

// ---------------- K2: scatter-add edges (16 lanes per edge) ----------------
__global__ __launch_bounds__(256) void scatter_edges_kernel(
        const float* __restrict__ edge_attr,   // [N_EDGES, EDGE_OUT]
        const int*   __restrict__ edge_index,  // [2, N_EDGES], row 0 = src
        float* __restrict__ sums,              // [N_NODES, EDGE_OUT]
        float* __restrict__ counts)            // [N_NODES]
{
    long long tid = (long long)blockIdx.x * blockDim.x + threadIdx.x;
    int e = (int)(tid >> 4);
    int f = ((int)tid & 15) * 4;
    if (e >= N_EDGES) return;
    int src = edge_index[e];
    // one-shot stream: non-temporal load keeps L2 for reused data
    v4f v = __builtin_nontemporal_load((const v4f*)(edge_attr + (size_t)e * EDGE_OUT + f));
    float* dst = sums + (size_t)src * EDGE_OUT + f;
    atomicAdd(dst + 0, v.x);
    atomicAdd(dst + 1, v.y);
    atomicAdd(dst + 2, v.z);
    atomicAdd(dst + 3, v.w);
    if ((tid & 15) == 0) atomicAdd(counts + src, 1.0f);
}

// ---------------- K3: fused concat + MLP (bf16 WMMA, f32 accumulate) ----------------
__global__ __launch_bounds__(256) void fused_mlp_kernel(
        const float* __restrict__ x,       // [N_NODES, NODE_IN]
        const float* __restrict__ sums,    // [N_NODES, EDGE_OUT]
        const float* __restrict__ counts,  // [N_NODES]
        const int*   __restrict__ batch,   // [N_NODES]
        const float* __restrict__ u,       // [N_GRAPHS, GLOBAL_IN]
        const __bf16* __restrict__ W1P,    // [KC1][HIDDEN][W_STRIDE]
        const float* __restrict__ b1,      // [HIDDEN]
        const __bf16* __restrict__ W2P,    // [KC2][NODE_OUT][W_STRIDE]
        const float* __restrict__ b2,      // [NODE_OUT]
        float* __restrict__ out)           // [N_NODES, NODE_OUT]
{
    extern __shared__ __bf16 smem[];
    __bf16* sA  = smem;                        // [M_TILE][A_STRIDE] combined input tile
    __bf16* sW0 = sA + M_TILE * A_STRIDE;      // weight chunk slot 0
    __bf16* sW1 = sW0 + CH1_ELEMS;             // weight chunk slot 1
    __bf16* sH  = sW1 + CH1_ELEMS;             // [M_TILE][H_STRIDE] hidden activations

    const int m0    = blockIdx.x * M_TILE;
    const int tid   = threadIdx.x;
    const int wave  = tid >> 5;
    const int lane  = tid & 31;
    const int lhalf = lane >> 4;    // 0: lanes 0-15, 1: lanes 16-31
    const int l16   = lane & 15;

    // Prologue: start streaming GEMM1 chunk 0 while we build the input tile
    copy16_issue(sW0, W1P, CH1_ELEMS / 8, tid);

    // ---- Phase 1: build combined bf16 tile [x | scatter_mean | u[batch]], float4-wide ----
    for (int i = tid; i < M_TILE * (IN_DIM / 4); i += 256) {
        int r  = i / (IN_DIM / 4);
        int c4 = (i - r * (IN_DIM / 4)) * 4;
        int node = m0 + r;
        v4f v = {0.f, 0.f, 0.f, 0.f};
        if (node < N_NODES) {
            if (c4 < NODE_IN) {
                v = __builtin_nontemporal_load((const v4f*)(x + (size_t)node * NODE_IN + c4));
            } else if (c4 < NODE_IN + EDGE_OUT) {
                float inv = 1.0f / fmaxf(counts[node], 1.0f);
                v4f s = *(const v4f*)(sums + (size_t)node * EDGE_OUT + (c4 - NODE_IN));
                v = s * inv;
            } else {
                int g = batch[node];
                v = *(const v4f*)(u + (size_t)g * GLOBAL_IN + (c4 - NODE_IN - EDGE_OUT));
            }
        }
        v4bf o = { (__bf16)v.x, (__bf16)v.y, (__bf16)v.z, (__bf16)v.w };
        *(v4bf*)(sA + r * A_STRIDE + c4) = o;
    }
    wait_async0();
    __syncthreads();

    // ---- Phase 2: GEMM1  h = relu(A[64x448] @ W1[448x512] + b1) ----
    // 4 row-tiles x 32 col-tiles = 128 tiles over 8 waves -> 16 tiles/wave
    v8f acc[16] = {};
    const int rt  = wave & 3;
    const int nt0 = (wave >> 2) * 16;
    // A-frag (ISA 16-bit 16x32): lane<16 -> K 0-7 & 16-23 of row l16; lane>=16 -> K 8-15 & 24-31
    const __bf16* aRow = sA + (rt * 16 + l16) * A_STRIDE + lhalf * 8;

    for (int kc = 0; kc < KC1; ++kc) {
        const __bf16* cur = (kc & 1) ? sW1 : sW0;
        if (kc + 1 < KC1)   // stream next chunk while computing on this one
            copy16_issue((kc & 1) ? sW0 : sW1,
                         W1P + (size_t)(kc + 1) * CH1_ELEMS, CH1_ELEMS / 8, tid);

        v16bf afrag = cat8(*(const v8bf*)(aRow + kc * 32),
                           *(const v8bf*)(aRow + kc * 32 + 16));
        v16bf bfrag = loadB(cur, nt0, l16, lhalf);
        #pragma unroll
        for (int j = 0; j < 16; ++j) {       // software-pipelined B-frag double buffer
            v16bf nb = (j < 15) ? loadB(cur, nt0 + j + 1, l16, lhalf) : bfrag;
            acc[j] = __builtin_amdgcn_wmma_f32_16x16x32_bf16(
                         false, afrag, false, bfrag, (short)0, acc[j], false, false);
            bfrag = nb;
        }
        if (kc + 1 < KC1) wait_async0();
        __syncthreads();
    }

    // Start streaming GEMM2 chunk 0; it overlaps the bias+ReLU spill below
    copy16_issue(sW0, W2P, CH2_ELEMS / 8, tid);

    // ---- Phase 3: bias + ReLU, spill h tile to LDS as bf16 ----
    #pragma unroll
    for (int j = 0; j < 16; ++j) {
        int col = (nt0 + j) * 16 + l16;
        float bias = b1[col];
        #pragma unroll
        for (int r = 0; r < 8; ++r) {
            int row = rt * 16 + lhalf * 8 + r;           // D layout: M = r + 8*lhalf
            float v = acc[j][r] + bias;
            sH[row * H_STRIDE + col] = (__bf16)fmaxf(v, 0.0f);
        }
    }
    wait_async0();
    __syncthreads();

    // ---- Phase 4: GEMM2  out = h[64x512] @ W2[512x256] + b2 ----
    // 4 row-tiles x 16 col-tiles = 64 tiles over 8 waves -> 8 tiles/wave
    v8f acc2[8] = {};
    const int rt2  = wave & 3;
    const int nt02 = (wave >> 2) * 8;
    const __bf16* hRow = sH + (rt2 * 16 + l16) * H_STRIDE + lhalf * 8;

    for (int kc = 0; kc < KC2; ++kc) {
        const __bf16* cur = (kc & 1) ? sW1 : sW0;
        if (kc + 1 < KC2)
            copy16_issue((kc & 1) ? sW0 : sW1,
                         W2P + (size_t)(kc + 1) * CH2_ELEMS, CH2_ELEMS / 8, tid);

        v16bf afrag = cat8(*(const v8bf*)(hRow + kc * 32),
                           *(const v8bf*)(hRow + kc * 32 + 16));
        v16bf bfrag = loadB(cur, nt02, l16, lhalf);
        #pragma unroll
        for (int j = 0; j < 8; ++j) {
            v16bf nb = (j < 7) ? loadB(cur, nt02 + j + 1, l16, lhalf) : bfrag;
            acc2[j] = __builtin_amdgcn_wmma_f32_16x16x32_bf16(
                          false, afrag, false, bfrag, (short)0, acc2[j], false, false);
            bfrag = nb;
        }
        if (kc + 1 < KC2) wait_async0();
        __syncthreads();
    }

    // ---- Phase 5: bias + non-temporal f32 output stores ----
    #pragma unroll
    for (int j = 0; j < 8; ++j) {
        int col = (nt02 + j) * 16 + l16;
        float bias = b2[col];
        #pragma unroll
        for (int r = 0; r < 8; ++r) {
            int row = rt2 * 16 + lhalf * 8 + r;
            int node = m0 + row;
            if (node < N_NODES)
                __builtin_nontemporal_store(acc2[j][r] + bias,
                                            out + (size_t)node * NODE_OUT + col);
        }
    }
}

// ---------------- launcher ----------------
extern "C" void kernel_launch(void* const* d_in, const int* in_sizes, int n_in,
                              void* d_out, int out_size, void* d_ws, size_t ws_size,
                              hipStream_t stream) {
    const float* x          = (const float*)d_in[0];
    const int*   edge_index = (const int*)  d_in[1];
    const float* edge_attr  = (const float*)d_in[2];
    const float* u          = (const float*)d_in[3];
    const int*   batch      = (const int*)  d_in[4];
    const float* W1         = (const float*)d_in[5];
    const float* b1         = (const float*)d_in[6];
    const float* W2         = (const float*)d_in[7];
    const float* b2         = (const float*)d_in[8];
    float* out = (float*)d_out;

    float*  sums   = (float*)d_ws;
    float*  counts = sums + SUMS_FLOATS;
    __bf16* W1P    = (__bf16*)((char*)d_ws + W1P_OFF_B);
    __bf16* W2P    = (__bf16*)((char*)d_ws + W2P_OFF_B);

    // K1: zero scatter accumulators (deterministic every call)
    int zeroN = N_NODES * (EDGE_OUT + 1);
    zero_ws_kernel<<<(zeroN + 255) / 256, 256, 0, stream>>>(sums, zeroN);

    // K0: pack weights -> bf16, LDS-ready transposed layout
    pack_weights_kernel<<<(IN_DIM * HIDDEN + 255) / 256, 256, 0, stream>>>(W1, W2, W1P, W2P);

    // K2: scatter-add edge features + counts
    long long sthreads = (long long)N_EDGES * 16;
    scatter_edges_kernel<<<(int)((sthreads + 255) / 256), 256, 0, stream>>>(
        edge_attr, edge_index, sums, counts);

    // K3: fused concat + 2-layer MLP with bf16 WMMA; double-buffered async weight stream
    size_t smem_bytes = (size_t)(M_TILE * A_STRIDE + 2 * CH1_ELEMS + M_TILE * H_STRIDE)
                        * sizeof(__bf16);   // ~207 KB of the 320 KB WGP LDS
    fused_mlp_kernel<<<(N_NODES + M_TILE - 1) / M_TILE, 256, smem_bytes, stream>>>(
        x, sums, counts, batch, u, W1P, b1, W2P, b2, out);
}
